// Bilstm_MTL_31482110279729
// MI455X (gfx1250) — compile-verified, matrix operand order checked by reading, not run on hardware
//
#include <hip/hip_runtime.h>
#include <hip/hip_bf16.h>
#include <math.h>

// ---------------- sizes ----------------
#define BB   32
#define SS   256
#define EE   300
#define EPAD 320          // E padded to multiple of 32 for WMMA K
#define HH   256
#define G4H  1024         // 4*H
#define TT   19           // tags incl START/STOP
#define TPAD 32
#define STARTT 17
#define STOPT  18
#define MM   (BB*SS)      // 8192 rows, m = s*32 + b

typedef __attribute__((ext_vector_type(16))) __bf16 v16bf;
typedef __attribute__((ext_vector_type(8)))  float  v8f;
typedef int b128i __attribute__((vector_size(16)));   // 16-byte payload for async LDS DMA

// ---------------- WMMA helpers (16x16x32 bf16, f32 acc) ----------------
__device__ __forceinline__ v8f wmma_bf16(v16bf a, v16bf b, v8f c) {
  return __builtin_amdgcn_wmma_f32_16x16x32_bf16(
      /*neg_a=*/false, a, /*neg_b=*/false, b,
      /*c_mod=*/(short)0, c, /*reuse_a=*/false, /*reuse_b=*/false);
}

// Load A or B operand tile (CDNA5 ISA 7.12.2 16-bit 16x32 layout).
__device__ __forceinline__ v16bf load_tile_ab(const __bf16* __restrict__ base,
                                              int stride, int k0, int lane) {
  const int r  = lane & 15;
  const int hi = (lane >> 4) << 3;          // 0 or 8
  const __bf16* p = base + (size_t)r * stride + k0 + hi;
  v16bf v;
#pragma unroll
  for (int q = 0; q < 8; ++q) {
    const int kb = (q < 4) ? (2 * q) : (2 * q + 8);
    v[2 * q]     = p[kb];
    v[2 * q + 1] = p[kb + 1];
  }
  return v;
}

__device__ __forceinline__ float sigmf(float x) { return 1.0f / (1.0f + expf(-x)); }

__device__ __forceinline__ void wait_async_zero() {
#if __has_builtin(__builtin_amdgcn_s_wait_asynccnt)
  __builtin_amdgcn_s_wait_asynccnt(0);
#else
  asm volatile("s_wait_asynccnt 0x0" ::: "memory");
#endif
}

// ---------------- kernel 1: weight prep (f32 -> bf16 layouts) ----------------
__global__ void prep_kernel(const float* __restrict__ Wih_f, const float* __restrict__ Whh_f,
                            const float* __restrict__ bih_f, const float* __restrict__ bhh_f,
                            const float* __restrict__ Wih_b, const float* __restrict__ Whh_b,
                            const float* __restrict__ bih_b, const float* __restrict__ bhh_b,
                            const float* __restrict__ Wtag,  const float* __restrict__ btag,
                            __bf16* __restrict__ WihB, __bf16* __restrict__ WhhB,
                            __bf16* __restrict__ WtagB, float* __restrict__ biasB,
                            float* __restrict__ btagB) {
  const int gsz = gridDim.x * blockDim.x;
  const int g0  = blockIdx.x * blockDim.x + threadIdx.x;
  for (int i = g0; i < 2 * G4H * EPAD; i += gsz) {
    const int d = i / (G4H * EPAD);
    const int r = (i / EPAD) % G4H;
    const int k = i % EPAD;
    const float* W = d ? Wih_b : Wih_f;
    WihB[i] = (__bf16)((k < EE) ? W[(size_t)r * EE + k] : 0.0f);
  }
  for (int i = g0; i < 2 * G4H * HH; i += gsz) {
    const int d = i / (G4H * HH);
    const float* W = d ? Whh_b : Whh_f;
    WhhB[i] = (__bf16)W[i - d * G4H * HH];
  }
  for (int i = g0; i < TPAD * (2 * HH); i += gsz) {
    const int n = i / (2 * HH);
    const int k = i % (2 * HH);
    WtagB[i] = (__bf16)((n < TT) ? Wtag[(size_t)n * (2 * HH) + k] : 0.0f);
  }
  for (int i = g0; i < 2 * G4H; i += gsz) {
    const int d = i / G4H;
    const int r = i % G4H;
    biasB[i] = d ? (bih_b[r] + bhh_b[r]) : (bih_f[r] + bhh_f[r]);
  }
  for (int i = g0; i < TPAD; i += gsz) btagB[i] = (i < TT) ? btag[i] : 0.0f;
}

// ---------------- kernel 2: embedding gather + mask -> bf16 [s][b][EPAD] ----
__global__ void embed_kernel(const int* __restrict__ word, const int* __restrict__ mask,
                             const float* __restrict__ emb, __bf16* __restrict__ X) {
  const int m = blockIdx.x;          // m = s*32 + b
  const int s = m >> 5;
  const int b = m & 31;
  const int e = threadIdx.x;         // 0..319
  const int w  = word[b * SS + s];
  const float mk = (float)mask[b * SS + s];
  const float v  = (e < EE) ? emb[(size_t)w * EE + e] * mk : 0.0f;
  X[(size_t)m * EPAD + e] = (__bf16)v;
}

// ---------------- kernel 3: input projection GEMM (4x N register blocking) --
// G[dir][m][n] = X[m][:] @ Wih[dir][n][:] + bias[dir][n]  (M=8192,N=1024,K=320)
__global__ void gemm_in_kernel(const __bf16* __restrict__ X, const __bf16* __restrict__ W,
                               const float* __restrict__ bias, float* __restrict__ G) {
  const int lane = threadIdx.x & 31;
  const int wg   = blockIdx.x * (blockDim.x >> 5) + (threadIdx.x >> 5); // 16384 jobs
  const int dir  = wg >> 13;                // 8192 jobs per direction
  const int rem  = wg & 8191;
  const int m0   = (rem >> 4) << 4;         // 512 m-tiles
  const int ng   = rem & 15;                // n-group: 4 tiles -> n0 = ng*64
  const int hi   = (lane >> 4) << 3;
  v8f acc[4];
#pragma unroll
  for (int u = 0; u < 4; ++u) {
    const float bsc = bias[dir * G4H + ng * 64 + u * 16 + (lane & 15)];
#pragma unroll
    for (int i = 0; i < 8; ++i) acc[u][i] = bsc;
  }
  const __bf16* A = X + (size_t)m0 * EPAD;
  const __bf16* B = W + ((size_t)dir * G4H + ng * 64) * EPAD;
#pragma unroll
  for (int kt = 0; kt < EPAD / 32; ++kt) {
    __builtin_prefetch(A + (kt + 1) * 32, 0, 1);
    const v16bf a = load_tile_ab(A, EPAD, kt * 32, lane);
#pragma unroll
    for (int u = 0; u < 4; ++u) {
      const v16bf b = load_tile_ab(B + (size_t)u * 16 * EPAD, EPAD, kt * 32, lane);
      acc[u] = wmma_bf16(a, b, acc[u]);
    }
  }
#pragma unroll
  for (int u = 0; u < 4; ++u) {
    float* out = G + ((size_t)dir * MM + m0 + hi) * G4H + ng * 64 + u * 16 + (lane & 15);
#pragma unroll
    for (int i = 0; i < 8; ++i) out[(size_t)i * G4H] = acc[u][i];
  }
}

// ---------------- kernel 4: persistent recurrent LSTM -----------------------
// Grid = 2 (one WG/direction), 512 threads = 16 waves. Wave w owns n-tiles
// {w, w+16, w+32, w+48} = gate slices i/f/g/o for hidden units [16w,16w+16):
// cell update happens elementwise on the WMMA accumulators (c in registers).
// Whh rows n<512 pinned in LDS (staged via async-to-LDS); n>=512 stream L2.
__global__ void lstm_kernel(const __bf16* __restrict__ Whh, const float* __restrict__ G,
                            __bf16* __restrict__ Hbf) {
  __shared__ __align__(16) __bf16 s_w[512 * HH];   // 256 KB: Whh rows [0,512)
  __shared__ __align__(16) __bf16 s_h[BB * HH];    //  16 KB: current h (bf16)
  const int dir  = blockIdx.x;
  const int tid  = threadIdx.x;
  const int w    = tid >> 5;           // wave 0..15
  const int lane = tid & 31;
  const int hi   = (lane >> 4) << 3;
  const float*  Gd  = G   + (size_t)dir * MM * G4H;
  const __bf16* Whd = Whh + (size_t)dir * G4H * HH;

  // ---- stage Whh[n<512] into LDS (async DMA path if available) ----
#if __has_builtin(__builtin_amdgcn_global_load_async_to_lds_b128)
  for (int i = tid * 8; i < 512 * HH; i += 512 * 8) {
    __builtin_amdgcn_global_load_async_to_lds_b128(
        (__attribute__((address_space(1))) b128i*)(Whd + i),
        (__attribute__((address_space(3))) b128i*)(&s_w[i]), 0, 0);
  }
  wait_async_zero();
#else
  for (int i = tid * 8; i < 512 * HH; i += 512 * 8) {
    *(float4*)&s_w[i] = *(const float4*)(Whd + i);
  }
#endif
  for (int u = tid; u < BB * HH; u += 512) s_h[u] = (__bf16)0.0f;
  __syncthreads();

  v8f cst[2];
#pragma unroll
  for (int mt = 0; mt < 2; ++mt)
#pragma unroll
    for (int i = 0; i < 8; ++i) cst[mt][i] = 0.0f;

  for (int step = 0; step < SS; ++step) {
    const int t = dir ? (SS - 1 - step) : step;
    // ---- gates = h @ Whh^T + G[t] for this wave's 8 tiles ----
    v8f acc[4][2];                         // [gate][mtile]
#pragma unroll
    for (int g = 0; g < 4; ++g) {
      const int n = g * 256 + (w << 4) + (lane & 15);
#pragma unroll
      for (int mt = 0; mt < 2; ++mt) {
        const float* g0 = Gd + (size_t)(t * BB + mt * 16 + hi) * G4H + n;
#pragma unroll
        for (int i = 0; i < 8; ++i) acc[g][mt][i] = g0[(size_t)i * G4H];
      }
    }
#pragma unroll
    for (int kt = 0; kt < HH / 32; ++kt) {
      const v16bf a0 = load_tile_ab(s_h, HH, kt * 32, lane);
      const v16bf a1 = load_tile_ab(s_h + 16 * HH, HH, kt * 32, lane);
#pragma unroll
      for (int g = 0; g < 4; ++g) {
        const int n0 = g * 256 + (w << 4);
        const __bf16* Bt = (g < 2) ? (const __bf16*)(s_w + (size_t)n0 * HH)
                                   : (Whd + (size_t)n0 * HH);
        if (g >= 2) __builtin_prefetch(Bt + (kt + 1) * 32, 0, 1);
        const v16bf b = load_tile_ab(Bt, HH, kt * 32, lane);
        acc[g][0] = wmma_bf16(a0, b, acc[g][0]);
        acc[g][1] = wmma_bf16(a1, b, acc[g][1]);
      }
    }
    __syncthreads();   // all waves done reading s_h of step t
    // ---- elementwise cell update in registers; publish h ----
    const int j = (w << 4) + (lane & 15);  // hidden index owned by this lane
#pragma unroll
    for (int mt = 0; mt < 2; ++mt) {
#pragma unroll
      for (int i = 0; i < 8; ++i) {
        const float ig = acc[0][mt][i], fg = acc[1][mt][i];
        const float gg = acc[2][mt][i], og = acc[3][mt][i];
        const float c  = sigmf(fg) * cst[mt][i] + sigmf(ig) * tanhf(gg);
        const float hv = sigmf(og) * tanhf(c);
        cst[mt][i] = c;
        const int b = mt * 16 + i + hi;    // batch index
        const __bf16 hb = (__bf16)hv;
        s_h[b * HH + j] = hb;
        Hbf[((size_t)(t * BB + b)) * (2 * HH) + dir * HH + j] = hb;
      }
    }
    __syncthreads();   // h published for step t+1
  }
}

// ---------------- kernel 5: tag projection GEMM -----------------------------
// logits[m][n] = Hcat[m][:512] @ Wtag[n][:] + btag[n]  (M=8192, N=32pad, K=512)
__global__ void gemm_tag_kernel(const __bf16* __restrict__ Hbf, const __bf16* __restrict__ W,
                                const float* __restrict__ btag, float* __restrict__ logits) {
  const int lane = threadIdx.x & 31;
  const int wg   = blockIdx.x * (blockDim.x >> 5) + (threadIdx.x >> 5);  // 1024 tiles
  const int m0   = (wg >> 1) << 4;
  const int n0   = (wg & 1) << 4;
  const int n    = n0 + (lane & 15);
  const int hi   = (lane >> 4) << 3;
  const float bsc = btag[n];
  v8f acc;
#pragma unroll
  for (int i = 0; i < 8; ++i) acc[i] = bsc;
  const __bf16* A = Hbf + (size_t)m0 * (2 * HH);
  const __bf16* B = W   + (size_t)n0 * (2 * HH);
#pragma unroll
  for (int kt = 0; kt < (2 * HH) / 32; ++kt) {
    v16bf a = load_tile_ab(A, 2 * HH, kt * 32, lane);
    v16bf b = load_tile_ab(B, 2 * HH, kt * 32, lane);
    acc = wmma_bf16(a, b, acc);
  }
  float* out = logits + (size_t)(m0 + hi) * TPAD + n;
#pragma unroll
  for (int i = 0; i < 8; ++i) out[(size_t)i * TPAD] = acc[i];
}

// ---------------- kernel 6: CRF NLL + Viterbi -------------------------------
__global__ void crf_kernel(const float* __restrict__ logits, const int* __restrict__ mask,
                           const int* __restrict__ labels, const float* __restrict__ trans,
                           int* __restrict__ ptrs, float* __restrict__ out) {
  __shared__ float tr[TT * TT];
  __shared__ float al[BB][33];
  __shared__ float dl[BB][33];
  __shared__ float lossb[BB];
  const int j = threadIdx.x, b = threadIdx.y;
  const int tid = b * 32 + j;
  for (int i = tid; i < TT * TT; i += 1024) tr[i] = trans[i];
  __syncthreads();
  float alpha = 0.0f, delta = 0.0f;
  if (j < TT) {
    const float e0 = logits[(size_t)(0 * BB + b) * TPAD + j];
    alpha = e0 + tr[STARTT * TT + j];
    delta = alpha;
  }
  for (int s = 1; s < SS; ++s) {
    al[b][j] = alpha; dl[b][j] = delta;
    __syncthreads();
    if (j < TT) {
      const int   m = mask[b * SS + s];
      const float e = logits[(size_t)(s * BB + b) * TPAD + j];
      float mx = -1e30f, mxd = -1e30f; int arg = 0;
      for (int p = 0; p < TT; ++p) {
        const float va = al[b][p] + tr[p * TT + j];
        if (va > mx) mx = va;
        const float vd = dl[b][p] + tr[p * TT + j];
        if (vd > mxd) { mxd = vd; arg = p; }
      }
      float ssum = 0.0f;
      for (int p = 0; p < TT; ++p) ssum += expf(al[b][p] + tr[p * TT + j] - mx);
      if (m > 0) {
        alpha = mx + logf(ssum) + e;
        delta = mxd + e;
        ptrs[(size_t)(s * BB + b) * TPAD + j] = arg;
      } else {
        ptrs[(size_t)(s * BB + b) * TPAD + j] = j;   // identity pass-through
      }
    }
    __syncthreads();
  }
  if (j < TT) {
    al[b][j] = alpha + tr[j * TT + STOPT];
    dl[b][j] = delta + tr[j * TT + STOPT];
  }
  __syncthreads();
  if (j == 0) {
    float mx = -1e30f;
    for (int p = 0; p < TT; ++p) mx = fmaxf(mx, al[b][p]);
    float ssum = 0.0f;
    for (int p = 0; p < TT; ++p) ssum += expf(al[b][p] - mx);
    const float logZ = mx + logf(ssum);
    float bmx = -1e30f; int barg = 0;
    for (int p = 0; p < TT; ++p) if (dl[b][p] > bmx) { bmx = dl[b][p]; barg = p; }
    float gold = 0.0f; int len = 0; int prev = STARTT;
    for (int s = 0; s < SS; ++s) {
      const int m   = mask[b * SS + s];
      const int lab = labels[b * SS + s];
      const float sc = tr[prev * TT + lab] + logits[(size_t)(s * BB + b) * TPAD + lab];
      gold += m ? sc : 0.0f;
      len  += m;
      prev  = lab;
    }
    const int last = labels[b * SS + (len - 1)];
    gold += tr[last * TT + STOPT];
    lossb[b] = logZ - gold;
    int cur = barg;
    for (int s = SS - 1; s >= 1; --s) {
      out[1 + b * SS + s] = (float)(mask[b * SS + s] ? cur : 0);
      cur = ptrs[(size_t)(s * BB + b) * TPAD + cur];
    }
    out[1 + b * SS + 0] = (float)(mask[b * SS + 0] ? cur : 0);
  }
  __syncthreads();
  if (tid == 0) {
    float tot = 0.0f;
    for (int p = 0; p < BB; ++p) tot += lossb[p];
    out[0] = tot / (float)BB;
  }
}

// ---------------- host launcher ----------------
static inline size_t alup(size_t x) { return (x + 255) & ~(size_t)255; }

extern "C" void kernel_launch(void* const* d_in, const int* in_sizes, int n_in,
                              void* d_out, int out_size, void* d_ws, size_t ws_size,
                              hipStream_t stream) {
  (void)in_sizes; (void)n_in; (void)out_size; (void)ws_size;
  const int*   word   = (const int*)d_in[0];
  const int*   mask   = (const int*)d_in[1];
  const int*   labels = (const int*)d_in[2];
  const float* emb    = (const float*)d_in[5];
  const float* Wih_f  = (const float*)d_in[6];
  const float* Whh_f  = (const float*)d_in[7];
  const float* bih_f  = (const float*)d_in[8];
  const float* bhh_f  = (const float*)d_in[9];
  const float* Wih_b  = (const float*)d_in[10];
  const float* Whh_b  = (const float*)d_in[11];
  const float* bih_b  = (const float*)d_in[12];
  const float* bhh_b  = (const float*)d_in[13];
  const float* Wtag   = (const float*)d_in[14];
  const float* btag   = (const float*)d_in[15];
  const float* trans  = (const float*)d_in[16];
  float* out = (float*)d_out;

  char* ws = (char*)d_ws;
  size_t off = 0;
  __bf16* Xbf   = (__bf16*)(ws + off); off = alup(off + (size_t)MM * EPAD * 2);
  __bf16* WihB  = (__bf16*)(ws + off); off = alup(off + (size_t)2 * G4H * EPAD * 2);
  __bf16* WhhB  = (__bf16*)(ws + off); off = alup(off + (size_t)2 * G4H * HH * 2);
  __bf16* WtagB = (__bf16*)(ws + off); off = alup(off + (size_t)TPAD * 2 * HH * 2);
  float*  biasB = (float*)(ws + off);  off = alup(off + (size_t)2 * G4H * 4);
  float*  btagB = (float*)(ws + off);  off = alup(off + (size_t)TPAD * 4);
  float*  G     = (float*)(ws + off);  off = alup(off + (size_t)2 * MM * G4H * 4);
  __bf16* Hbf   = (__bf16*)(ws + off); off = alup(off + (size_t)MM * 2 * HH * 2);
  float*  logit = (float*)(ws + off);  off = alup(off + (size_t)MM * TPAD * 4);
  int*    ptrs  = (int*)(ws + off);    off = alup(off + (size_t)MM * TPAD * 4);

  prep_kernel<<<512, 256, 0, stream>>>(Wih_f, Whh_f, bih_f, bhh_f,
                                       Wih_b, Whh_b, bih_b, bhh_b,
                                       Wtag, btag, WihB, WhhB, WtagB, biasB, btagB);
  embed_kernel<<<MM, EPAD, 0, stream>>>(word, mask, emb, Xbf);
  gemm_in_kernel<<<2048, 256, 0, stream>>>(Xbf, WihB, biasB, G);   // 16384 wave jobs
  lstm_kernel<<<2, 512, 0, stream>>>(WhhB, G, Hbf);
  gemm_tag_kernel<<<128, 256, 0, stream>>>(Hbf, WtagB, btagB, logit); // 1024 tiles
  crf_kernel<<<1, dim3(32, 32), 0, stream>>>(logit, mask, labels, trans, ptrs, out);
}
